// WindowAttention_21638045237964
// MI455X (gfx1250) — matrix operand order repaired and checked by reading, not compiled, not run on hardware
//
#include <hip/hip_runtime.h>

// ---------------------------------------------------------------------------
// Fused Swin window attention for MI455X (gfx1250), f16 WMMA / f32 accumulate.
//   kernel 1: qkv_w f32 -> f16 into d_ws (lives in 192MB L2, reused by all blocks)
//   kernel 2: one 256-thread block per 7x7 window:
//             x->LDS(f16) -> qkv^T = W @ x^T (wmma, vectorized q/k stores) ->
//             scores^T = k @ q^T (wmma, vectorized stores) -> softmax ->
//             out = attn @ v (wmma) -> coalesced NCHW store
// GEMMs are oriented so C-tile lanes hold 8 *consecutive* elements along the
// LDS-contiguous axis, turning the epilogues into 16/32-byte vector DS stores.
// ---------------------------------------------------------------------------

typedef _Float16 v16h __attribute__((ext_vector_type(16)));
typedef _Float16 v8h  __attribute__((ext_vector_type(8)));
typedef float    v8f  __attribute__((ext_vector_type(8)));
typedef float    v4f  __attribute__((ext_vector_type(4)));

#define SHUF16(lo, hi) __builtin_shufflevector(lo, hi, 0,1,2,3,4,5,6,7,8,9,10,11,12,13,14,15)

// Problem constants
#define BATCH   8
#define CDIM    384
#define HDIM    224
#define WDIM    224
#define NH      4
#define HD      96
#define TOK     49     // 7*7 tokens per window
#define TP      64     // padded tokens (4 tiles of 16)
#define N3      1152   // 3*CDIM
#define NWIN    32     // windows per side

// LDS layout (bytes). Peak 268288 B, single workgroup per WGP (320KB LDS).
#define Q_OFF   0                       // [4][64][104] f16 = 53248
#define K_OFF   53248                   // [4][64][104] f16 = 53248
#define V_OFF   106496                  // [4][96][72]  f16 = 55296 (transposed: [d][tok])
#define X_OFF   161792                  // [64][392]    f16 = 50176   (phase A only)
#define S_OFF   161792                  // [4][64][68]  f32 = 69632   (phase B, overlays X)
#define A_OFF   231424                  // [4][64][72]  f16 = 36864   (phase B)
#define O_OFF   161792                  // [384][52]    f32 = 79872   (phase B tail, overlays S)
#define SMEM_BYTES 268288

#define QK_ROW  104      // f16 elems per q/k row (96 + 8 pad)
#define QK_HEAD (64*QK_ROW)
#define V_ROW   72       // f16 elems per v row (64 + 8 pad)
#define V_HEAD  (96*V_ROW)
#define X_ROW   392      // f16 elems per x row (384 + 8 pad)
#define S_ROW   68       // f32 elems per score row
#define S_HEAD  (64*S_ROW)
#define A_ROW   72       // f16 elems per attn row
#define A_HEAD  (64*A_ROW)
#define O_ROW   52       // f32 elems per out-stage row

// A-fragment (16x32 f16): lane holds row=lane%16, K = k0..k0+15
// (k0 = 16*(lane>=16)) -> two contiguous 16B reads.
static __device__ __forceinline__ v16h load_a_frag(const _Float16* p) {
  v8h lo = *(const v8h*)p;
  v8h hi = *(const v8h*)(p + 8);
  return SHUF16(lo, hi);
}
// B-fragment (32x16 f16): lane%16 = N column, lane/16 selects K half; 16 contiguous K
// per lane -> one contiguous 32B read (two 16B vectors).
static __device__ __forceinline__ v16h load_b_frag(const _Float16* p) {
  v8h lo = *(const v8h*)p;
  v8h hi = *(const v8h*)(p + 8);
  return SHUF16(lo, hi);
}

static __device__ __forceinline__ v8f wmma_f16(v16h a, v16h b, v8f c) {
  return __builtin_amdgcn_wmma_f32_16x16x32_f16(false, a, false, b, (short)0, c, false, false);
}

// ---------------------------------------------------------------------------
__global__ void wconvert_f32_to_f16(const float* __restrict__ w,
                                    _Float16* __restrict__ wf16, int n) {
  int i = blockIdx.x * 256 + threadIdx.x;
  if (i < n) wf16[i] = (_Float16)w[i];
}

// ---------------------------------------------------------------------------
__global__ __launch_bounds__(256)
void fused_window_attention(const float* __restrict__ x,
                            const _Float16* __restrict__ wf16,   // [1152][384] f16
                            const float* __restrict__ bias,      // [1152]
                            float* __restrict__ out) {
  __shared__ alignas(16) unsigned char smem[SMEM_BYTES];
  _Float16* qS = (_Float16*)(smem + Q_OFF);
  _Float16* kS = (_Float16*)(smem + K_OFF);
  _Float16* vS = (_Float16*)(smem + V_OFF);
  _Float16* xS = (_Float16*)(smem + X_OFF);
  float*    sS = (float*)   (smem + S_OFF);
  _Float16* aS = (_Float16*)(smem + A_OFF);
  float*    oS = (float*)   (smem + O_OFF);

  const int tid  = threadIdx.x;
  const int lane = tid & 31;
  const int wv   = tid >> 5;
  const int lm   = lane & 15;   // fragment row/col lane coordinate
  const int lh   = lane >> 4;   // fragment K-half / M-half selector

  const int win = blockIdx.x;
  const int b   = win >> 10;          // window / 1024
  const int wy  = (win >> 5) & 31;
  const int wx  = win & 31;
  const int h0  = wy * 7, w0 = wx * 7;
  const size_t imgBase = (size_t)b * CDIM * HDIM * WDIM;

  // ---- stage x window into LDS as f16 ([tok][chan]) ----
  for (int e = tid; e < TOK * CDIM; e += 256) {
    int c = e / TOK, t = e - c * TOK;
    int ty = t / 7, tx = t - ty * 7;
    float v = x[imgBase + ((size_t)c * HDIM + (h0 + ty)) * WDIM + (w0 + tx)];
    xS[t * X_ROW + c] = (_Float16)v;
  }
  for (int e = tid; e < (TP - TOK) * CDIM; e += 256) {   // zero pad rows 49..63
    int t = TOK + e / CDIM, c = e % CDIM;
    xS[t * X_ROW + c] = (_Float16)0.f;
  }
  __syncthreads();

  // ============ Phase A: qkv^T = W @ x^T  (+bias) =========================
  // D[n][tok]: M-dim = output channel n, N-dim = token. Wave wv owns
  // n in [wv*144, wv*144+144) = 9 M-tiles, all 4 token tiles.
  // Register blocking 3(M) x 4(N), K inner (12 chunks of 32).
  // A = W rows streamed from L2 (read exactly once per block),
  // B = x from LDS. C-tile: lane holds fixed token, 8 consecutive n ->
  // q/k epilogue is a single 16B vector DS store per tile.
  {
    const int nwave = wv * 144;
    for (int ng = 0; ng < 3; ++ng) {
      v8f acc[3][4];
#pragma unroll
      for (int i = 0; i < 3; ++i)
#pragma unroll
        for (int j = 0; j < 4; ++j) acc[i][j] = (v8f){0.f,0.f,0.f,0.f,0.f,0.f,0.f,0.f};

      for (int kc = 0; kc < 12; ++kc) {
        const int ka = kc * 32 + lh * 16;   // A segment base
        const int kb = kc * 32 + lh * 16;   // B segment base
        v16h afr[3];
#pragma unroll
        for (int i = 0; i < 3; ++i) {
          int n = nwave + (ng * 3 + i) * 16 + lm;       // A row = channel n
          afr[i] = load_a_frag(wf16 + (size_t)n * 384 + ka);
        }
        v16h bfr[4];
#pragma unroll
        for (int j = 0; j < 4; ++j) {
          int m = j * 16 + lm;                          // B column = token
          bfr[j] = load_b_frag(xS + m * X_ROW + kb);
        }
#pragma unroll
        for (int i = 0; i < 3; ++i)
#pragma unroll
          for (int j = 0; j < 4; ++j)
            acc[i][j] = wmma_f16(afr[i], bfr[j], acc[i][j]);
      }

      // Epilogue: bias + store. Lane holds n-run [n0, n0+8) (8-aligned, never
      // crosses a 96/384 boundary), token = j*16+lm.
#pragma unroll
      for (int i = 0; i < 3; ++i) {
        const int n0  = nwave + (ng * 3 + i) * 16 + lh * 8;
        const int sec = n0 / 384;            // 0=q 1=k 2=v
        const int cch = n0 - sec * 384;
        const int hh2 = cch / HD;
        const int d0  = cch - hh2 * HD;
        const v4f b0 = *(const v4f*)(bias + n0);
        const v4f b1 = *(const v4f*)(bias + n0 + 4);
#pragma unroll
        for (int j = 0; j < 4; ++j) {
          const int m = j * 16 + lm;         // token
          v8h hv;
#pragma unroll
          for (int e = 0; e < 4; ++e) hv[e]     = (_Float16)(acc[i][j][e]     + b0[e]);
#pragma unroll
          for (int e = 0; e < 4; ++e) hv[4 + e] = (_Float16)(acc[i][j][4 + e] + b1[e]);
          if (sec == 0) {
            *(v8h*)(qS + hh2 * QK_HEAD + m * QK_ROW + d0) = hv;     // 16B store
          } else if (sec == 1) {
            *(v8h*)(kS + hh2 * QK_HEAD + m * QK_ROW + d0) = hv;     // 16B store
          } else {
#pragma unroll
            for (int e = 0; e < 8; ++e)                              // v: [d][tok]
              vS[hh2 * V_HEAD + (d0 + e) * V_ROW + m] = hv[e];
          }
        }
      }
    }
  }
  __syncthreads();

  // ================= Phase B: attention ===================================
  const int hh   = wv >> 1;        // head (2 waves per head)
  const int half = wv & 1;
  const float scale = 0.10206207261596577f;  // 96^-0.5

  // ---- scores^T = k @ q^T (K = 96, 3 chunks). D[key n][query m]:
  // lane holds fixed query m, 8 consecutive key n -> 32B vector stores
  // into sS[query][key] (the layout softmax wants).
  {
    v8f acc[4][2];                 // [key n-tile][query m-tile]
#pragma unroll
    for (int nt = 0; nt < 4; ++nt)
#pragma unroll
      for (int j = 0; j < 2; ++j) acc[nt][j] = (v8f){0.f,0.f,0.f,0.f,0.f,0.f,0.f,0.f};

    for (int kc = 0; kc < 3; ++kc) {
      const int ka = kc * 32 + lh * 16;
      const int kb = kc * 32 + lh * 16;
      v16h afr[4];
#pragma unroll
      for (int nt = 0; nt < 4; ++nt)     // A row = key token
        afr[nt] = load_a_frag(kS + hh * QK_HEAD + (nt * 16 + lm) * QK_ROW + ka);
      v16h bfr[2];
#pragma unroll
      for (int j = 0; j < 2; ++j) {      // B column = query token
        int m = (half * 2 + j) * 16 + lm;
        bfr[j] = load_b_frag(qS + hh * QK_HEAD + m * QK_ROW + kb);
      }
#pragma unroll
      for (int nt = 0; nt < 4; ++nt)
#pragma unroll
        for (int j = 0; j < 2; ++j)
          acc[nt][j] = wmma_f16(afr[nt], bfr[j], acc[nt][j]);
    }
#pragma unroll
    for (int nt = 0; nt < 4; ++nt)
#pragma unroll
      for (int j = 0; j < 2; ++j) {
        const int m  = (half * 2 + j) * 16 + lm;   // query token (lane-fixed)
        const int n0 = nt * 16 + lh * 8;           // 8 consecutive key tokens
        float* dst = sS + hh * S_HEAD + m * S_ROW + n0;
        v4f s0, s1;
#pragma unroll
        for (int e = 0; e < 4; ++e) { s0[e] = acc[nt][j][e] * scale;
                                      s1[e] = acc[nt][j][4 + e] * scale; }
        *(v4f*)dst       = s0;
        *(v4f*)(dst + 4) = s1;
      }
  }
  __syncthreads();

  // ---- softmax over 49 key tokens (rows are contiguous); attn -> f16 ----
  if (tid < NH * TOK) {
    int sh = tid / TOK, m = tid - sh * TOK;
    float* row = sS + sh * S_HEAD + m * S_ROW;
    float mx = row[0];
    for (int n = 1; n < TOK; ++n) mx = fmaxf(mx, row[n]);
    float sum = 0.f;
    for (int n = 0; n < TOK; ++n) { float e = __expf(row[n] - mx); row[n] = e; sum += e; }
    float inv = 1.f / sum;
    _Float16* arow = aS + sh * A_HEAD + m * A_ROW;
    for (int n = 0; n < TOK; ++n) arow[n] = (_Float16)(row[n] * inv);
    for (int n = TOK; n < TP; ++n) arow[n] = (_Float16)0.f;   // kill K-padding
  }
  __syncthreads();

  // ---- out = attn @ v  (K = 64 tokens, 2 chunks; N = 96 head dims) ----
  v8f oacc[4][3];
#pragma unroll
  for (int mt = 0; mt < 4; ++mt)
#pragma unroll
    for (int j = 0; j < 3; ++j) oacc[mt][j] = (v8f){0.f,0.f,0.f,0.f,0.f,0.f,0.f,0.f};

  for (int kc = 0; kc < 2; ++kc) {
    const int ka = kc * 32 + lh * 16;
    const int kb = kc * 32 + lh * 16;
    v16h afr[4];
#pragma unroll
    for (int mt = 0; mt < 4; ++mt)       // A row = query token
      afr[mt] = load_a_frag(aS + hh * A_HEAD + (mt * 16 + lm) * A_ROW + ka);
    v16h bfr[3];
#pragma unroll
    for (int j = 0; j < 3; ++j) {        // B column = head dim (v transposed)
      int d = half * 48 + j * 16 + lm;
      bfr[j] = load_b_frag(vS + hh * V_HEAD + d * V_ROW + kb);
    }
#pragma unroll
    for (int mt = 0; mt < 4; ++mt)
#pragma unroll
      for (int j = 0; j < 3; ++j)
        oacc[mt][j] = wmma_f16(afr[mt], bfr[j], oacc[mt][j]);
  }
  __syncthreads();   // all attn/v reads done -> safe to overlay out-stage

  // ---- stage out [c][t] f32: lane holds fixed d, 8 consecutive tokens ----
#pragma unroll
  for (int mt = 0; mt < 4; ++mt)
#pragma unroll
    for (int j = 0; j < 3; ++j) {
      const int d   = half * 48 + j * 16 + lm;
      const int cch = hh * HD + d;
      const int m0  = mt * 16 + lh * 8;
      if (mt < 3) {                                   // fully in-range: 32B store
        v4f o0, o1;
#pragma unroll
        for (int e = 0; e < 4; ++e) { o0[e] = oacc[mt][j][e]; o1[e] = oacc[mt][j][4 + e]; }
        float* dst = oS + cch * O_ROW + m0;
        *(v4f*)dst       = o0;
        *(v4f*)(dst + 4) = o1;
      } else {                                        // partial tile (m 48..63)
#pragma unroll
        for (int e = 0; e < 8; ++e) {
          int m = m0 + e;
          if (m < TOK) oS[cch * O_ROW + m] = oacc[mt][j][e];
        }
      }
    }
  __syncthreads();

  for (int e = tid; e < CDIM * TOK; e += 256) {
    int cch = e / TOK, t = e - cch * TOK;
    int ty = t / 7, tx = t - ty * 7;
    out[imgBase + ((size_t)cch * HDIM + (h0 + ty)) * WDIM + (w0 + tx)] = oS[cch * O_ROW + t];
  }
}

// ---------------------------------------------------------------------------
extern "C" void kernel_launch(void* const* d_in, const int* in_sizes, int n_in,
                              void* d_out, int out_size, void* d_ws, size_t ws_size,
                              hipStream_t stream) {
  const float* x      = (const float*)d_in[0];   // [8,384,224,224]
  const float* qkv_w  = (const float*)d_in[1];   // [1152,384]
  const float* qkv_b  = (const float*)d_in[2];   // [1152]
  float*       out    = (float*)d_out;
  _Float16*    wf16   = (_Float16*)d_ws;         // 884736 bytes of scratch

  const int wn = N3 * CDIM;                      // 442368 weights
  wconvert_f32_to_f16<<<(wn + 255) / 256, 256, 0, stream>>>(qkv_w, wf16, wn);

  const int nwin = BATCH * NWIN * NWIN;          // 8192 windows
  fused_window_attention<<<nwin, 256, 0, stream>>>(x, wf16, qkv_b, out);
}
